// PCC_encoders_17892833755635
// MI455X (gfx1250) — matrix operand order repaired and checked by reading, not compiled, not run on hardware
//
#include <hip/hip_runtime.h>

typedef __attribute__((ext_vector_type(16))) _Float16 v16h;
typedef __attribute__((ext_vector_type(8)))  _Float16 v8h;
typedef __attribute__((ext_vector_type(8)))  float    v8f;
typedef __attribute__((ext_vector_type(4)))  float    f32x4;
typedef __attribute__((ext_vector_type(4)))  _Float16 h16x4;

#define B_  131072
#define K_  512
#define N_  1024
#define BM  64
#define BN  64
#define LDS_STRIDE (K_ + 8)   // halves; +16B row pad to rotate LDS banks

struct H16Pair { v8h lo, hi; };

__global__ __launch_bounds__(128, 1)
void gf2_encode_wmma(const float* __restrict__ X,
                     const float* __restrict__ G,
                     float* __restrict__ Out) {
  __shared__ __align__(16) _Float16 ldsA[BM * LDS_STRIDE];
  __shared__ __align__(16) _Float16 ldsB[BN * LDS_STRIDE];

  const int tid  = threadIdx.x;
  const int lane = tid & 31;          // wave32
  const int wave = tid >> 5;          // 0..3
  const size_t r0 = (size_t)blockIdx.x * BM;

  // ---- Phase 1: stage A slab (f32 -> f16) into LDS, read X exactly once ----
  {
    const int f4PerRow = K_ / 4;      // 128 float4 per row
    for (int i = tid; i < BM * f4PerRow; i += 128) {
      const int m  = i / f4PerRow;
      const int kk = (i % f4PerRow) * 4;
      f32x4 v = *(const f32x4*)(X + (r0 + (size_t)m) * K_ + kk);
      h16x4 h;
      h.x = (_Float16)v.x; h.y = (_Float16)v.y;
      h.z = (_Float16)v.z; h.w = (_Float16)v.w;
      *(h16x4*)(&ldsA[m * LDS_STRIDE + kk]) = h;
    }
  }
  __syncthreads();

  // ---- Hoist this wave's 16 A fragments (full K=512) into registers ----
  // A 16x32 f16 layout (ISA 7.12.2): lane l, row m=l&15,
  //   halves[0..7]  = K = kb8..kb8+7        (kb8 = (l>>4)*8)
  //   halves[8..15] = K = 16+kb8..16+kb8+7
  const int m0   = wave * 16;
  const int mrow = m0 + (lane & 15);
  const int kb8  = (lane >> 4) * 8;
  v16h afrag[16];
#pragma unroll
  for (int kk = 0; kk < 16; ++kk) {
    const int k0 = kk * 32;
    H16Pair p;
    p.lo = *(const v8h*)(&ldsA[mrow * LDS_STRIDE + k0 + kb8]);
    p.hi = *(const v8h*)(&ldsA[mrow * LDS_STRIDE + k0 + 16 + kb8]);
    afrag[kk] = __builtin_bit_cast(v16h, p);
  }

  // B 32x16 f16 layout: lane l, col n=l&15, halves[0..7] = K = kb8..kb8+7, halves[8..15] = K = 16+kb8..16+kb8+7
  const int kb16 = (lane >> 4) * 16;
  const int ncol = lane & 15;

  // ---- Sweep all N in chunks of BN; G chunks come from L2 (G is 2MB) ----
#pragma unroll 1
  for (int nc = 0; nc < N_ / BN; ++nc) {
    const size_t n0 = (size_t)nc * BN;
    __syncthreads();                  // prior chunk's reads of ldsB are done
    {
      const int f4PerRow = K_ / 4;
      for (int i = tid; i < BN * f4PerRow; i += 128) {
        const int n  = i / f4PerRow;
        const int kk = (i % f4PerRow) * 4;
        f32x4 v = *(const f32x4*)(G + (n0 + (size_t)n) * K_ + kk);
        h16x4 h;
        h.x = (_Float16)v.x; h.y = (_Float16)v.y;
        h.z = (_Float16)v.z; h.w = (_Float16)v.w;
        *(h16x4*)(&ldsB[n * LDS_STRIDE + kk]) = h;
      }
    }
    __syncthreads();

#pragma unroll
    for (int nt = 0; nt < BN / 16; ++nt) {
      v8f acc = {};
      const int brow = nt * 16 + ncol;
#pragma unroll
      for (int kk = 0; kk < 16; ++kk) {
        const int k0 = kk * 32;
        H16Pair p;
        p.lo = *(const v8h*)(&ldsB[brow * LDS_STRIDE + k0 + kb8]);
        p.hi = *(const v8h*)(&ldsB[brow * LDS_STRIDE + k0 + 16 + kb8]);
        v16h bfrag = __builtin_bit_cast(v16h, p);
        acc = __builtin_amdgcn_wmma_f32_16x16x32_f16(
            /*neg_a=*/false, afrag[kk], /*neg_b=*/false, bfrag,
            /*c_mod=*/(short)0, acc, /*reuse_a=*/false, /*reuse_b=*/false);
      }
      // C/D layout: VGPR v, lane l -> row m = v + 8*(l>>4), col n = l&15.
      // Accumulator is an exact integer in [0,512]; mod 2 == low bit.
      const size_t rowBase = r0 + m0 + 8 * (size_t)(lane >> 4);
      const size_t col = n0 + (size_t)(nt * 16 + ncol);
#pragma unroll
      for (int v = 0; v < 8; ++v) {
        const int iv = (int)acc[v];
        Out[(rowBase + v) * N_ + col] = (float)(iv & 1);
      }
    }
  }
}

extern "C" void kernel_launch(void* const* d_in, const int* in_sizes, int n_in,
                              void* d_out, int out_size, void* d_ws, size_t ws_size,
                              hipStream_t stream) {
  (void)in_sizes; (void)n_in; (void)d_ws; (void)ws_size; (void)out_size;
  const float* X = (const float*)d_in[0];   // [B, K] f32 (0/1 values)
  const float* G = (const float*)d_in[1];   // [N, K] f32 (0/1 values)
  float* Out = (float*)d_out;               // [B, N] f32
  dim3 grid(B_ / BM);
  gf2_encode_wmma<<<grid, 128, 0, stream>>>(X, G, Out);
}
